// LSTM_decoder_89137751261809
// MI455X (gfx1250) — matrix operand order repaired and checked by reading, not compiled, not run on hardware
//
#include <hip/hip_runtime.h>
#include <cstdint>
#include <cstddef>

// ---------------------------------------------------------------------------
// MI455X (gfx1250) implementation: bf16 WMMA everywhere, persistent
// cooperative scan kernel with global software barriers, time-parallel
// injection/projection pre-pass.  All bulk memory ops forced to the GLOBAL
// address space (global_load_b128 etc.) so they only use LOADcnt/STOREcnt.
// ---------------------------------------------------------------------------

typedef __bf16 bf16;
typedef __attribute__((ext_vector_type(16))) __bf16 v16bf;
typedef __attribute__((ext_vector_type(8)))  __bf16 v8bf;
typedef __attribute__((ext_vector_type(8)))  float  v8f;

#define AS_GLOBAL __attribute__((address_space(1)))

#define T_LEN 256
#define BATCH 64
#define DDIM  512
#define HDIM  512
#define NTAGS 32
#define LM_K  200
#define LMP   224   // padded K for lm embedding
#define SMP   32    // padded K for pos/cap/poscap embeddings

__device__ __forceinline__ float sigmoidf_(float x) { return 1.0f / (1.0f + __expf(-x)); }

// ---- global-address-space scalar/vector access helpers --------------------
__device__ __forceinline__ float gldf(const float* p) { return *(const AS_GLOBAL float*)p; }
__device__ __forceinline__ void  gstf(float* p, float v) { *(AS_GLOBAL float*)p = v; }
__device__ __forceinline__ bf16  gldb(const bf16* p) { return *(const AS_GLOBAL bf16*)p; }
__device__ __forceinline__ void  gstb(bf16* p, bf16 v) { *(AS_GLOBAL bf16*)p = v; }

__device__ __forceinline__ v8f wmma_bf16(v16bf a, v16bf b, v8f c) {
  // v_wmma_f32_16x16x32_bf16  D = A(16x32) * B(32x16) + C
  return __builtin_amdgcn_wmma_f32_16x16x32_bf16(false, a, false, b, (short)0, c, false, false);
}

// A fragment: 16x32 bf16 tile from row-major [M][K] at (row0, k0).
// A layout used here: lane(0..15)=row, halves hold K 0..15 / 16..31 (contiguous, as in load_bT).
// kStride must be a multiple of 8 elements; k0 a multiple of 32.
__device__ __forceinline__ v16bf load_a(const bf16* __restrict__ src, int kStride,
                                        int row0, int k0) {
  const int lane = threadIdx.x & 31;
  const int row  = row0 + (lane & 15);
  const int half = lane >> 4;
  const bf16* p = src + (size_t)row * kStride + k0 + 16 * half;
  v8bf lo = *(const AS_GLOBAL v8bf*)(p);
  v8bf hi = *(const AS_GLOBAL v8bf*)(p + 8);
  v16bf f;
#pragma unroll
  for (int i = 0; i < 8; ++i) { f[i] = lo[i]; f[8 + i] = hi[i]; }
  return f;
}

// B fragment: 32x16 bf16 tile; source stored K-contiguous ("transposed"):
// srcT is [N][K] row-major. lane(0..15)=col, half selects K 0..15 / 16..31.
// Single aligned 32-byte vector load per lane.
__device__ __forceinline__ v16bf load_bT(const bf16* __restrict__ srcT, int kStride,
                                         int k0, int col0) {
  const int lane = threadIdx.x & 31;
  const int col  = col0 + (lane & 15);
  const int half = lane >> 4;
  return *(const AS_GLOBAL v16bf*)(srcT + (size_t)col * kStride + k0 + 16 * half);
}

// C/D tile (16x16 f32) stored to a K-contiguous bf16 activation buffer:
// element r of the accumulator is row row0+r+8*half, column col0+(lane&15).
__device__ __forceinline__ void store_cT(bf16* __restrict__ dstT, int kStride,
                                         int row0, int col0, v8f acc) {
  const int lane = threadIdx.x & 31;
  const int col  = col0 + (lane & 15);
  const int half = lane >> 4;
  v8bf o;
#pragma unroll
  for (int r = 0; r < 8; ++r) o[r] = (bf16)acc[r];
  *(AS_GLOBAL v8bf*)(dstT + (size_t)col * kStride + row0 + 8 * half) = o;
}

// ---------------------------------------------------------------------------
// Conversion kernels
// ---------------------------------------------------------------------------

// dst[m][k] (dims M x Kp, bf16) = src (optionally transposed) zero-padded in K.
__global__ void cvt_w_pad(const float* __restrict__ src, bf16* __restrict__ dst,
                          int M, int K, int Kp, int trans) {
  int idx = blockIdx.x * blockDim.x + threadIdx.x;
  if (idx >= M * Kp) return;
  int m = idx / Kp, k = idx % Kp;
  float v = 0.0f;
  if (k < K) v = trans ? gldf(src + (size_t)k * M + m) : gldf(src + (size_t)m * K + k);
  gstb(dst + idx, (bf16)v);
}

// src [T][L][B] f32  ->  dst [T][B][Lp] bf16 (K-contiguous, zero padded)
__global__ void cvt_act_T(const float* __restrict__ src, bf16* __restrict__ dst,
                          int L, int Lp) {
  int idx = blockIdx.x * blockDim.x + threadIdx.x;
  int total = T_LEN * BATCH * Lp;
  if (idx >= total) return;
  int t = idx / (BATCH * Lp);
  int rem = idx % (BATCH * Lp);
  int b = rem / Lp, l = rem % Lp;
  float v = (l < L) ? gldf(src + ((size_t)t * L + l) * BATCH + b) : 0.0f;
  gstb(dst + idx, (bf16)v);
}

// ---------------------------------------------------------------------------
// Injection pre-pass:  X = x + sum_g proj_g * sigmoid(mg_g^T @ x + b_g)
// one wave per (t, 16-row tile, 16-col tile); fully fused, no cross-wave deps.
// ---------------------------------------------------------------------------

struct InjArgs {
  const float* x;                        // [T][D][B] f32 (original inputs)
  const bf16*  XinT;                     // [T][B][D] bf16(inputs)
  const bf16  *MgT0, *MgT1, *MgT2, *MgT3;   // [D][D] (mg transposed)
  const float *mgb0, *mgb1, *mgb2, *mgb3;   // [D]
  const bf16  *Ww0, *Ww1, *Ww2, *Ww3;    // [D][Lp] (w_*w transposed, padded)
  const bf16  *E0, *E1, *E2, *E3;        // [T][B][Lp]
  bf16* XT;                              // out: [T][B][D]
};

__global__ __launch_bounds__(256) void inject_kernel(InjArgs A) {
  const int gw   = blockIdx.x * 8 + (threadIdx.x >> 5);
  const int t    = gw >> 7;
  const int rem  = gw & 127;
  const int m    = rem >> 2;
  const int n    = (rem & 3) << 4;
  const int lane = threadIdx.x & 31;
  const int half = lane >> 4;
  const int colL = n + (lane & 15);
  const int rbase = m * 16 + 8 * half;

  const bf16*  MgT[4] = {A.MgT0, A.MgT1, A.MgT2, A.MgT3};
  const float* mgb[4] = {A.mgb0, A.mgb1, A.mgb2, A.mgb3};
  const bf16*  Ww[4]  = {A.Ww0, A.Ww1, A.Ww2, A.Ww3};
  const int    Lp[4]  = {LMP, SMP, SMP, SMP};
  const bf16*  Eb[4]  = {A.E0 + (size_t)t * BATCH * LMP,
                         A.E1 + (size_t)t * BATCH * SMP,
                         A.E2 + (size_t)t * BATCH * SMP,
                         A.E3 + (size_t)t * BATCH * SMP};
  const bf16* Xt = A.XinT + (size_t)t * BATCH * DDIM;

  v8f xacc;
#pragma unroll
  for (int r = 0; r < 8; ++r)
    xacc[r] = gldf(A.x + ((size_t)t * DDIM + rbase + r) * BATCH + colL);

#pragma unroll
  for (int g = 0; g < 4; ++g) {
    v8f ga = {};
    for (int k = 0; k < DDIM; k += 32)
      ga = wmma_bf16(load_a(MgT[g], DDIM, m * 16, k), load_bT(Xt, DDIM, k, n), ga);
    v8f pa = {};
    for (int k = 0; k < Lp[g]; k += 32)
      pa = wmma_bf16(load_a(Ww[g], Lp[g], m * 16, k), load_bT(Eb[g], Lp[g], k, n), pa);
#pragma unroll
    for (int r = 0; r < 8; ++r)
      xacc[r] += pa[r] * sigmoidf_(ga[r] + gldf(mgb[g] + rbase + r));
  }
  store_cT(A.XT + (size_t)t * BATCH * DDIM, DDIM, m * 16, n, xacc);
}

// ---------------------------------------------------------------------------
// Tag-side projection pre-pass:  P_g[t] = w_*t @ emb_t   ([H][B] per t)
// ---------------------------------------------------------------------------

struct ProjArgs {
  const bf16 *W0, *W1, *W2, *W3;  // [H][Lp]
  const bf16 *E0, *E1, *E2, *E3;  // [T][B][Lp]
  bf16* PT;                       // out: [4][T][B][H]
};

__global__ __launch_bounds__(256) void proj_kernel(ProjArgs A) {
  const int gw  = blockIdx.x * 8 + (threadIdx.x >> 5);
  const int t   = gw >> 7;
  const int rem = gw & 127;
  const int m   = rem >> 2;
  const int n   = (rem & 3) << 4;

  const bf16* W[4]  = {A.W0, A.W1, A.W2, A.W3};
  const int   Lp[4] = {LMP, SMP, SMP, SMP};
  const bf16* Eb[4] = {A.E0 + (size_t)t * BATCH * LMP,
                       A.E1 + (size_t)t * BATCH * SMP,
                       A.E2 + (size_t)t * BATCH * SMP,
                       A.E3 + (size_t)t * BATCH * SMP};
#pragma unroll
  for (int g = 0; g < 4; ++g) {
    v8f pa = {};
    for (int k = 0; k < Lp[g]; k += 32)
      pa = wmma_bf16(load_a(W[g], Lp[g], m * 16, k), load_bT(Eb[g], Lp[g], k, n), pa);
    store_cT(A.PT + ((size_t)g * T_LEN + t) * BATCH * HDIM, HDIM, m * 16, n, pa);
  }
}

// ---------------------------------------------------------------------------
// Persistent cooperative scan kernel.
// Grid = 16 blocks x 256 threads = 128 waves = 32 row-tiles x 4 col-tiles.
// Global barrier between the GEMM phases of each timestep.
// ---------------------------------------------------------------------------

struct ScanArgs {
  const bf16 *Wii, *Whi, *Wti, *Wco, *Wht;  // bf16 weights, [M][512]
  const bf16 *Tg0, *Tg1, *Tg2, *Tg3;        // [H][H]
  const bf16 *P0, *P1, *P2, *P3;            // [T][B][H] precomputed projections
  const float *tgb0, *tgb1, *tgb2, *tgb3;   // [H]
  const float *b_i;                         // [5H]
  const bf16 *X;                            // [T][B][D] injected inputs
  bf16 *c2T, *hT, *TbT, *TnT;               // [B][H] activation state (K-contig)
  float *Tnf;                               // [B][H] f32 Tn for the softmax head
  const float *wyf, *byf, *wfact, *wyc, *byc;
  float *out;
  unsigned *bar;
};

__device__ __forceinline__ void gbar(unsigned* ctr, unsigned& target) {
  __threadfence();
  __syncthreads();
  if (threadIdx.x == 0) {
    __hip_atomic_fetch_add((AS_GLOBAL unsigned*)ctr, 1u, __ATOMIC_ACQ_REL,
                           __HIP_MEMORY_SCOPE_AGENT);
    const unsigned tgt = target;
    while (__hip_atomic_load((AS_GLOBAL unsigned*)ctr, __ATOMIC_ACQUIRE,
                             __HIP_MEMORY_SCOPE_AGENT) < tgt)
      __builtin_amdgcn_s_sleep(8);
  }
  __syncthreads();
  __threadfence();
  target += gridDim.x;
}

__global__ __launch_bounds__(256) void scan_kernel(ScanArgs A) {
  __shared__ float sY[2][NTAGS][BATCH];   // 16 KB: y_fact / y_cond staging

  const int tid   = threadIdx.x;
  const int gw    = blockIdx.x * 8 + (tid >> 5);   // 0..127
  const int m     = gw >> 2;                        // row tile 0..31
  const int n     = (gw & 3) << 4;                  // col0
  const int lane  = tid & 31;
  const int half  = lane >> 4;
  const int colL  = n + (lane & 15);
  const int rbase = m * 16 + 8 * half;

  const bf16*  Tg[4]  = {A.Tg0, A.Tg1, A.Tg2, A.Tg3};
  const bf16*  P[4]   = {A.P0, A.P1, A.P2, A.P3};
  const float* tgb[4] = {A.tgb0, A.tgb1, A.tgb2, A.tgb3};

  unsigned target = gridDim.x;

  // zero-init recurrent state h, T
  for (int i = blockIdx.x * 256 + tid; i < BATCH * HDIM; i += gridDim.x * 256) {
    gstb(A.hT + i, (bf16)0.0f);
    gstb(A.TnT + i, (bf16)0.0f);
  }
  v8f cReg = {};   // cell state: lives in registers (this wave owns its tile)
  gbar(A.bar, target);

  for (int t = 0; t < T_LEN; ++t) {
    const bf16* Xt = A.X + (size_t)t * BATCH * DDIM;

    // ---- P1: gi = Wii@x_t + Whi@h, ti = Wti@T ; fused i/f/z/c2 ----
    v8f g0 = {}, g1 = {}, g2 = {}, g3 = {}, t0 = {}, t1 = {}, t2 = {};
    for (int k = 0; k < DDIM; k += 32) {
      __builtin_prefetch(A.Whi + (size_t)(m * 16) * HDIM + k, 0, 3);
      v16bf b = load_bT(Xt, DDIM, k, n);
      g0 = wmma_bf16(load_a(A.Wii, DDIM, 0 * HDIM + m * 16, k), b, g0);
      g1 = wmma_bf16(load_a(A.Wii, DDIM, 1 * HDIM + m * 16, k), b, g1);
      g2 = wmma_bf16(load_a(A.Wii, DDIM, 2 * HDIM + m * 16, k), b, g2);
      g3 = wmma_bf16(load_a(A.Wii, DDIM, 3 * HDIM + m * 16, k), b, g3);
    }
    for (int k = 0; k < HDIM; k += 32) {
      v16bf b = load_bT(A.hT, HDIM, k, n);
      g0 = wmma_bf16(load_a(A.Whi, HDIM, 0 * HDIM + m * 16, k), b, g0);
      g1 = wmma_bf16(load_a(A.Whi, HDIM, 1 * HDIM + m * 16, k), b, g1);
      g2 = wmma_bf16(load_a(A.Whi, HDIM, 2 * HDIM + m * 16, k), b, g2);
      g3 = wmma_bf16(load_a(A.Whi, HDIM, 3 * HDIM + m * 16, k), b, g3);
    }
    for (int k = 0; k < HDIM; k += 32) {
      v16bf b = load_bT(A.TnT, HDIM, k, n);
      t0 = wmma_bf16(load_a(A.Wti, HDIM, 0 * HDIM + m * 16, k), b, t0);
      t1 = wmma_bf16(load_a(A.Wti, HDIM, 1 * HDIM + m * 16, k), b, t1);
      t2 = wmma_bf16(load_a(A.Wti, HDIM, 2 * HDIM + m * 16, k), b, t2);
    }
    v8f goReg = g3;  // o-gate pre-activation (without w_co part), kept in regs
#pragma unroll
    for (int r = 0; r < 8; ++r) {
      const int row = rbase + r;
      float iv = sigmoidf_(g0[r] + t0[r] + gldf(A.b_i + row));
      float fv = sigmoidf_(g1[r] + t1[r] + gldf(A.b_i + HDIM + row));
      float zv = tanhf(g2[r] + t2[r] + gldf(A.b_i + 2 * HDIM + row));
      cReg[r] = fv * cReg[r] + iv * zv;
    }
    store_cT(A.c2T, HDIM, m * 16, n, cReg);
    gbar(A.bar, target);

    // ---- P3: o = sigmoid(gi3 + Wco@c2 + b), h2 = o*tanh(c2) ----
    v8f oa = {};
    for (int k = 0; k < HDIM; k += 32)
      oa = wmma_bf16(load_a(A.Wco, HDIM, m * 16, k), load_bT(A.c2T, HDIM, k, n), oa);
    {
      v8f h2;
#pragma unroll
      for (int r = 0; r < 8; ++r) {
        const int row = rbase + r;
        float ov = sigmoidf_(goReg[r] + oa[r] + gldf(A.b_i + 3 * HDIM + row));
        h2[r] = ov * tanhf(cReg[r]);
      }
      store_cT(A.hT, HDIM, m * 16, n, h2);
    }
    gbar(A.bar, target);

    // ---- P4: Tb = Wht@h2 + b ----
    v8f tb = {};
    for (int k = 0; k < HDIM; k += 32)
      tb = wmma_bf16(load_a(A.Wht, HDIM, m * 16, k), load_bT(A.hT, HDIM, k, n), tb);
#pragma unroll
    for (int r = 0; r < 8; ++r) tb[r] += gldf(A.b_i + 4 * HDIM + rbase + r);
    store_cT(A.TbT, HDIM, m * 16, n, tb);
    gbar(A.bar, target);

    // ---- P5: Tn = Tb + sum_g P_g * sigmoid(Tg_g@Tb + b_g) ; emit hid ----
    v8f tn = tb;
#pragma unroll
    for (int g = 0; g < 4; ++g) {
      v8f s = {};
      for (int k = 0; k < HDIM; k += 32)
        s = wmma_bf16(load_a(Tg[g], HDIM, m * 16, k), load_bT(A.TbT, HDIM, k, n), s);
      v8bf pv = *(const AS_GLOBAL v8bf*)(P[g] + (size_t)t * BATCH * HDIM +
                                         (size_t)colL * HDIM + m * 16 + 8 * half);
#pragma unroll
      for (int r = 0; r < 8; ++r)
        tn[r] += (float)pv[r] * sigmoidf_(s[r] + gldf(tgb[g] + rbase + r));
    }
    store_cT(A.TnT, HDIM, m * 16, n, tn);
    {
      float* tf  = A.Tnf + (size_t)colL * HDIM + m * 16 + 8 * half;
      float* hid = A.out + (size_t)4 * BATCH * T_LEN * NTAGS +
                   (size_t)colL * T_LEN * HDIM + (size_t)t * HDIM + m * 16 + 8 * half;
#pragma unroll
      for (int r = 0; r < 8; ++r) { gstf(tf + r, tn[r]); gstf(hid + r, tn[r]); }
    }
    gbar(A.bar, target);

    // ---- P6 (workgroup 0 only): TAGS=32 softmax head, f32 ----
    if (blockIdx.x == 0) {
      const int b  = tid & 63;
      const int ty = tid >> 6;   // 0..3
      for (int ii = 0; ii < 8; ++ii) {
        const int i = ty * 8 + ii;
        float af = gldf(A.byf + i), ac = gldf(A.byc + i);
        const float* tn_ = A.Tnf + (size_t)b * HDIM;
        const float* wf  = A.wyf + (size_t)i * HDIM;
        const float* wc  = A.wyc + (size_t)i * HDIM;
        for (int k = 0; k < HDIM; ++k) {
          float tv = gldf(tn_ + k);
          af += gldf(wf + k) * tv;
          ac += gldf(wc + k) * tv;
        }
        sY[0][i][b] = af;
        sY[1][i][b] = ac;
      }
      __syncthreads();
      if (tid < BATCH) {
        const size_t R = (size_t)BATCH * T_LEN * NTAGS;
        float* lf = A.out + 0 * R + (size_t)tid * T_LEN * NTAGS + (size_t)t * NTAGS;
        float* lc = A.out + 1 * R + (size_t)tid * T_LEN * NTAGS + (size_t)t * NTAGS;
        float* df = A.out + 2 * R + (size_t)tid * T_LEN * NTAGS + (size_t)t * NTAGS;
        float* dc = A.out + 3 * R + (size_t)tid * T_LEN * NTAGS + (size_t)t * NTAGS;

        float mx = -1e30f;
#pragma unroll
        for (int i = 0; i < NTAGS; ++i) mx = fmaxf(mx, sY[0][i][tid]);
        float se = 0.0f;
#pragma unroll
        for (int i = 0; i < NTAGS; ++i) se += __expf(sY[0][i][tid] - mx);
        const float lse = mx + __logf(se);
        float sf[NTAGS];
#pragma unroll
        for (int i = 0; i < NTAGS; ++i) {
          float l = sY[0][i][tid] - lse;
          gstf(lf + i, l);
          float s = __expf(l);
          sf[i] = s;
          gstf(df + i, s);
        }
        float yc[NTAGS];
        float mx2 = -1e30f;
#pragma unroll
        for (int i = 0; i < NTAGS; ++i) {
          float a = sY[1][i][tid];
          const float* wr = A.wfact + (size_t)i * NTAGS;
#pragma unroll
          for (int j = 0; j < NTAGS; ++j) a += gldf(wr + j) * sf[j];
          yc[i] = a;
          mx2 = fmaxf(mx2, a);
        }
        float se2 = 0.0f;
#pragma unroll
        for (int i = 0; i < NTAGS; ++i) se2 += __expf(yc[i] - mx2);
        const float lse2 = mx2 + __logf(se2);
#pragma unroll
        for (int i = 0; i < NTAGS; ++i) {
          float l = yc[i] - lse2;
          gstf(lc + i, l);
          gstf(dc + i, __expf(l));
        }
      }
      __syncthreads();
    }
  }
}

// ---------------------------------------------------------------------------
// Host launch
// ---------------------------------------------------------------------------

extern "C" void kernel_launch(void* const* d_in, const int* in_sizes, int n_in,
                              void* d_out, int out_size, void* d_ws, size_t ws_size,
                              hipStream_t stream) {
  (void)in_sizes; (void)n_in; (void)out_size; (void)ws_size;

  char* ws = (char*)d_ws;
  size_t off = 0;
  auto take = [&](size_t bytes) -> char* {
    char* p = ws + off;
    off += (bytes + 255) & ~(size_t)255;
    return p;
  };

  unsigned* bar = (unsigned*)take(256);
  bf16* Wii    = (bf16*)take((size_t)2048 * 512 * 2);
  bf16* Whi    = (bf16*)take((size_t)2048 * 512 * 2);
  bf16* Wti    = (bf16*)take((size_t)1536 * 512 * 2);
  bf16* Wco    = (bf16*)take((size_t)512 * 512 * 2);
  bf16* Wht    = (bf16*)take((size_t)512 * 512 * 2);
  bf16* TgB    = (bf16*)take((size_t)4 * 512 * 512 * 2);
  bf16* MgTB   = (bf16*)take((size_t)4 * 512 * 512 * 2);
  bf16* WlmwT  = (bf16*)take((size_t)512 * LMP * 2);
  bf16* WposwT = (bf16*)take((size_t)512 * SMP * 2);
  bf16* WcapwT = (bf16*)take((size_t)512 * SMP * 2);
  bf16* WpcwT  = (bf16*)take((size_t)512 * SMP * 2);
  bf16* Wlmt   = (bf16*)take((size_t)512 * LMP * 2);
  bf16* Wpost  = (bf16*)take((size_t)512 * SMP * 2);
  bf16* Wcapt  = (bf16*)take((size_t)512 * SMP * 2);
  bf16* Wpct   = (bf16*)take((size_t)512 * SMP * 2);
  bf16* XinT   = (bf16*)take((size_t)T_LEN * BATCH * DDIM * 2);
  bf16* ElmT   = (bf16*)take((size_t)T_LEN * BATCH * LMP * 2);
  bf16* EposT  = (bf16*)take((size_t)T_LEN * BATCH * SMP * 2);
  bf16* EcapT  = (bf16*)take((size_t)T_LEN * BATCH * SMP * 2);
  bf16* EpcT   = (bf16*)take((size_t)T_LEN * BATCH * SMP * 2);
  bf16* XT     = (bf16*)take((size_t)T_LEN * BATCH * DDIM * 2);
  bf16* PT     = (bf16*)take((size_t)4 * T_LEN * BATCH * HDIM * 2);
  bf16* c2T    = (bf16*)take((size_t)BATCH * HDIM * 2);
  bf16* hT     = (bf16*)take((size_t)BATCH * HDIM * 2);
  bf16* TbT    = (bf16*)take((size_t)BATCH * HDIM * 2);
  bf16* TnT    = (bf16*)take((size_t)BATCH * HDIM * 2);
  float* Tnf   = (float*)take((size_t)BATCH * HDIM * 4);

  hipMemsetAsync(bar, 0, 256, stream);

  auto cvtW = [&](const void* src, bf16* dst, int M, int K, int Kp, int trans) {
    int nel = M * Kp;
    cvt_w_pad<<<(nel + 255) / 256, 256, 0, stream>>>((const float*)src, dst, M, K, Kp, trans);
  };
  // plain (row-major) weights
  cvtW(d_in[5], Wii, 2048, 512, 512, 0);
  cvtW(d_in[6], Whi, 2048, 512, 512, 0);
  cvtW(d_in[7], Wti, 1536, 512, 512, 0);
  cvtW(d_in[8], Wco, 512, 512, 512, 0);
  cvtW(d_in[9], Wht, 512, 512, 512, 0);
  for (int g = 0; g < 4; ++g) cvtW(d_in[24 + g], TgB + (size_t)g * 512 * 512, 512, 512, 512, 0);
  // transposed weights (einsum contractions use W^T as the WMMA A operand)
  for (int g = 0; g < 4; ++g) cvtW(d_in[16 + g], MgTB + (size_t)g * 512 * 512, 512, 512, 512, 1);
  cvtW(d_in[32], WlmwT,  512, LM_K, LMP, 1);
  cvtW(d_in[33], WposwT, 512, 5,    SMP, 1);
  cvtW(d_in[34], WcapwT, 512, 3,    SMP, 1);
  cvtW(d_in[35], WpcwT,  512, 6,    SMP, 1);
  cvtW(d_in[36], Wlmt,   512, LM_K, LMP, 0);
  cvtW(d_in[37], Wpost,  512, 5,    SMP, 0);
  cvtW(d_in[38], Wcapt,  512, 3,    SMP, 0);
  cvtW(d_in[39], Wpct,   512, 6,    SMP, 0);

  auto cvtA = [&](const void* src, bf16* dst, int L, int Lp) {
    int nel = T_LEN * BATCH * Lp;
    cvt_act_T<<<(nel + 255) / 256, 256, 0, stream>>>((const float*)src, dst, L, Lp);
  };
  cvtA(d_in[0], XinT,  512, 512);
  cvtA(d_in[1], ElmT,  LM_K, LMP);
  cvtA(d_in[2], EposT, 5, SMP);
  cvtA(d_in[3], EcapT, 3, SMP);
  cvtA(d_in[4], EpcT,  6, SMP);

  InjArgs ia;
  ia.x = (const float*)d_in[0];
  ia.XinT = XinT;
  ia.MgT0 = MgTB; ia.MgT1 = MgTB + (size_t)1 * 512 * 512;
  ia.MgT2 = MgTB + (size_t)2 * 512 * 512; ia.MgT3 = MgTB + (size_t)3 * 512 * 512;
  ia.mgb0 = (const float*)d_in[20]; ia.mgb1 = (const float*)d_in[21];
  ia.mgb2 = (const float*)d_in[22]; ia.mgb3 = (const float*)d_in[23];
  ia.Ww0 = WlmwT; ia.Ww1 = WposwT; ia.Ww2 = WcapwT; ia.Ww3 = WpcwT;
  ia.E0 = ElmT; ia.E1 = EposT; ia.E2 = EcapT; ia.E3 = EpcT;
  ia.XT = XT;
  inject_kernel<<<(T_LEN * 128) / 8, 256, 0, stream>>>(ia);

  ProjArgs pa;
  pa.W0 = Wlmt; pa.W1 = Wpost; pa.W2 = Wcapt; pa.W3 = Wpct;
  pa.E0 = ElmT; pa.E1 = EposT; pa.E2 = EcapT; pa.E3 = EpcT;
  pa.PT = PT;
  proj_kernel<<<(T_LEN * 128) / 8, 256, 0, stream>>>(pa);

  ScanArgs sa;
  sa.Wii = Wii; sa.Whi = Whi; sa.Wti = Wti; sa.Wco = Wco; sa.Wht = Wht;
  sa.Tg0 = TgB; sa.Tg1 = TgB + (size_t)1 * 512 * 512;
  sa.Tg2 = TgB + (size_t)2 * 512 * 512; sa.Tg3 = TgB + (size_t)3 * 512 * 512;
  sa.P0 = PT;                                    sa.P1 = PT + (size_t)1 * T_LEN * BATCH * HDIM;
  sa.P2 = PT + (size_t)2 * T_LEN * BATCH * HDIM; sa.P3 = PT + (size_t)3 * T_LEN * BATCH * HDIM;
  sa.tgb0 = (const float*)d_in[28]; sa.tgb1 = (const float*)d_in[29];
  sa.tgb2 = (const float*)d_in[30]; sa.tgb3 = (const float*)d_in[31];
  sa.b_i = (const float*)d_in[10];
  sa.X = XT;
  sa.c2T = c2T; sa.hT = hT; sa.TbT = TbT; sa.TnT = TnT; sa.Tnf = Tnf;
  sa.wyf = (const float*)d_in[11]; sa.byf = (const float*)d_in[12];
  sa.wfact = (const float*)d_in[13];
  sa.wyc = (const float*)d_in[14]; sa.byc = (const float*)d_in[15];
  sa.out = (float*)d_out;
  sa.bar = bar;
  scan_kernel<<<16, 256, 0, stream>>>(sa);
}
